// Bottleneck_11416023073044
// MI455X (gfx1250) — matrix operand order repaired and checked by reading, not compiled, not run on hardware
//
#include <hip/hip_runtime.h>
#include <hip/hip_bf16.h>
#include <math.h>

typedef __attribute__((ext_vector_type(16))) _Float16 v16h;
typedef __attribute__((ext_vector_type(8)))  float    v8f;
typedef _Float16 h16;

#define EPS 1e-5f
#define Bsz 8
#define C1 256
#define C2 256
#define Cm 128          // C_ = C2/2
#define H 80
#define W 80
#define HW 6400

// ---------------- workspace layout (bytes) ----------------
#define WS_H16     0            // h (post cv1) as f16: 8*128*6400*2 = 13,107,200
#define WS_A1PACK  13107200     // W1' packed A-fragments: 8kc*8mt*32*16*2 = 65,536
#define WS_A2PACK  13172736     // Wc' packed A-fragments: 36kc*16mt*32*16*2 = 589,824
#define WS_WGP     13762560     // Wg' f32: 128*81*4 = 41,472
#define WS_BGP     13804032     // bg' f32: 128*9*4 = 4,608
#define WS_C1B     13808640     // c1 bias f32: 128*4 (padded)
#define WS_BCP     13809152     // bc' f32: 256*4 (padded)

// B-tile row stride in halfs (K=576 per half + 16 pad; 1184 B rows keep v16h
// loads 32B-aligned and spread LDS banks)
#define BT_STRIDE 592
// halo tile per-channel stride: 4 rows x 18 cols
#define HS_STRIDE 72

// WMMA A fragment (16x32 f16) index helper: given lane(0..31) and packed half
// position p(0..15), returns k within the 32-wide K chunk; m = lane%16.
__device__ __forceinline__ int frag_k(int lane, int p) {
    return ((lane >= 16) ? 16 : 0) + p;
}

// ---------------- prep: fold BN into biases ----------------
__global__ void k_prep_bias(const float* g1, const float* b1, const float* m1, const float* v1,
                            const float* bc, const float* g2, const float* b2, const float* m2,
                            const float* v2, float* c1b, float* bcp) {
    int i = blockIdx.x * blockDim.x + threadIdx.x;
    if (i < Cm) {
        float a = g1[i] * rsqrtf(v1[i] + EPS);
        c1b[i] = b1[i] - a * m1[i];
    } else if (i < Cm + C2) {
        int m = i - Cm;
        float a = g2[m] * rsqrtf(v2[m] + EPS);
        bcp[m] = a * (bc[m] - m2[m]) + b2[m];
    }
}

// Wg' = ag*Wg ; bg' = ag*(bg - mg) + bgw   (ag = gg*rsqrt(vg+eps))
__global__ void k_prep_wg(const float* Wg, const float* bg, const float* gg, const float* bgw,
                          const float* mg, const float* vg, float* WgP, float* bgP) {
    int idx = blockIdx.x * blockDim.x + threadIdx.x;
    if (idx >= Cm * 90) return;
    int c = idx / 90, r = idx % 90;
    if (r < 81) {
        int o = r / 9;
        float ag = gg[c * 9 + o] * rsqrtf(vg[c * 9 + o] + EPS);
        WgP[c * 81 + r] = ag * Wg[c * 81 + r];
    } else {
        int o = r - 81;
        float ag = gg[c * 9 + o] * rsqrtf(vg[c * 9 + o] + EPS);
        bgP[c * 9 + o] = ag * (bg[c * 9 + o] - mg[c * 9 + o]) + bgw[c * 9 + o];
    }
}

// Pack W1' (BN-scaled) into A fragment layout: [kc(8)][mt(8)][lane(32)][16]
__global__ void k_prep_a1(const float* W1, const float* g1, const float* v1, h16* A1) {
    int idx = blockIdx.x * blockDim.x + threadIdx.x;          // 32768 total
    int p = idx & 15, lane = (idx >> 4) & 31, mt = (idx >> 9) & 7, kc = idx >> 12;
    int m = mt * 16 + (lane & 15);
    int k = kc * 32 + frag_k(lane, p);
    float a = g1[m] * rsqrtf(v1[m] + EPS);
    A1[idx] = (h16)(a * W1[m * C1 + k]);
}

// Pack Wc' into A fragment layout: [kc(36)][mt(16)][lane(32)][16]; K order k = c*9 + n
// (c-major, exactly the Wc memory order: Wc[m][c][n])
__global__ void k_prep_a2(const float* Wc, const float* g2, const float* v2, h16* A2) {
    int idx = blockIdx.x * blockDim.x + threadIdx.x;          // 294912 total
    int p = idx & 15, lane = (idx >> 4) & 31, mt = (idx >> 9) & 15, kc = idx >> 13;
    int m = mt * 16 + (lane & 15);
    int k = kc * 32 + frag_k(lane, p);
    float a = g2[m] * rsqrtf(v2[m] + EPS);
    A2[idx] = (h16)(a * Wc[m * (Cm * 9) + k]);
}

// ---------------- kernel 1: cv1 GEMM + BN + SiLU -> h f16 ----------------
// M=128, K=256, N=51200. Block = 256 thr (8 waves). Tile: M=128 x N=64 pixels.
__global__ void __launch_bounds__(256) k_cv1(const float* __restrict__ x,
                                             const h16* __restrict__ A1,
                                             const float* __restrict__ c1b,
                                             h16* __restrict__ hOut) {
    __shared__ __align__(32) h16 Bb[64 * 48];                 // [pix][chan], padded stride
    int b    = blockIdx.x / 100;
    int pix0 = (blockIdx.x % 100) * 64;
    int tid = threadIdx.x, lane = tid & 31, wave = tid >> 5;
    int wm = wave & 3, wn = wave >> 2;                        // 4 M-groups x 2 N-groups

    v8f acc[2][2] = {{{}, {}}, {{}, {}}};

    for (int kc = 0; kc < 8; ++kc) {
        __syncthreads();
        // stage x[b, kc*32 .. +31, pix0 .. +63] -> f16 LDS transposed
        #pragma unroll
        for (int i = 0; i < 8; ++i) {
            int e = tid + 256 * i;                            // 2048 elements
            int pix = e & 63, ch = e >> 6;
            float v = x[(size_t)(b * C1 + kc * 32 + ch) * HW + pix0 + pix];
            Bb[pix * 48 + ch] = (h16)v;
        }
        __syncthreads();

        v16h a0 = *(const v16h*)(A1 + ((size_t)(kc * 8 + 2 * wm)     * 32 + lane) * 16);
        v16h a1 = *(const v16h*)(A1 + ((size_t)(kc * 8 + 2 * wm + 1) * 32 + lane) * 16);
        #pragma unroll
        for (int nt = 0; nt < 2; ++nt) {
            const h16* bp = &Bb[(32 * wn + 16 * nt + (lane & 15)) * 48 + ((lane < 16) ? 0 : 16)];
            v16h bf = *(const v16h*)bp;
            acc[0][nt] = __builtin_amdgcn_wmma_f32_16x16x32_f16(false, a0, false, bf,
                                                                (short)0, acc[0][nt], false, false);
            acc[1][nt] = __builtin_amdgcn_wmma_f32_16x16x32_f16(false, a1, false, bf,
                                                                (short)0, acc[1][nt], false, false);
        }
    }

    // epilogue: bias + SiLU -> f16
    #pragma unroll
    for (int mt = 0; mt < 2; ++mt)
        #pragma unroll
        for (int nt = 0; nt < 2; ++nt)
            #pragma unroll
            for (int r = 0; r < 8; ++r) {
                int m = (2 * wm + mt) * 16 + 8 * (lane >> 4) + r;
                int pix = pix0 + 32 * wn + 16 * nt + (lane & 15);
                float v = acc[mt][nt][r] + c1b[m];
                v = v / (1.f + __expf(-v));                   // SiLU
                hOut[(size_t)(b * Cm + m) * HW + pix] = (h16)v;
            }
}

// ---------------- kernel 2: fused unfold + attention softmax + GEMM ----------------
// One block per (b, 2-row y tile, 16-wide x tile): N = 32 pixels, M = 256.
// K = 1152 (c-major: k = c*9+n), processed as 2 halves of 576 so the B tile
// fits in LDS alongside the halo tile.
__global__ void __launch_bounds__(256) k_rfa(const h16* __restrict__ hIn,
                                             const h16* __restrict__ A2,
                                             const float* __restrict__ WgP,
                                             const float* __restrict__ bgP,
                                             const float* __restrict__ bcp,
                                             const float* __restrict__ xin,
                                             float* __restrict__ out) {
    __shared__ __align__(32) h16 hs[Cm * HS_STRIDE];          // [c][4 rows][18 cols] halo
    __shared__ __align__(32) h16 Bt[32 * BT_STRIDE];          // [pix(32)][K half = 576]

    int blk = blockIdx.x;
    int b  = blk / 200;
    int rm = blk % 200;
    int y0 = (rm / 5) * 2;
    int x0 = (rm % 5) * 16;
    int tid = threadIdx.x, lane = tid & 31, wave = tid >> 5;

    // Phase A: load halo'd h tile: rows y0-1 .. y0+2, cols x0-1 .. x0+16 (zero pad)
    for (int e = tid; e < Cm * HS_STRIDE; e += 256) {
        int c = e / HS_STRIDE, r = e % HS_STRIDE, i = r / 18, xl = r % 18;
        int gy = y0 - 1 + i, gx = x0 - 1 + xl;
        h16 v = (h16)0.f;
        if (gy >= 0 && gy < H && gx >= 0 && gx < W)
            v = hIn[(size_t)(b * Cm + c) * HW + gy * W + gx];
        hs[e] = v;
    }

    v8f acc[2][2] = {{{}, {}}, {{}, {}}};
    int m0t = wave * 2;

    for (int half = 0; half < 2; ++half) {
        __syncthreads();   // hs ready / previous GEMM reads of Bt done

        // Phase B: per (c, pixel): 9 logits, softmax, weighted patches -> B tile.
        // c is wave-uniform -> Wg'/bg' reads are broadcast loads.
        {
            int pix = tid & 31;                               // 0..31: py = pix/16, px = pix%16
            int py = pix >> 4, px = pix & 15;
            int cb = tid >> 5;                                // 0..7 (wave id)
            #pragma unroll
            for (int cc = 0; cc < 8; ++cc) {
                int c = half * 64 + cb * 8 + cc;
                float p[9];
                #pragma unroll
                for (int i = 0; i < 3; ++i)
                    #pragma unroll
                    for (int j = 0; j < 3; ++j)
                        p[i * 3 + j] = (float)hs[c * HS_STRIDE + (py + i) * 18 + px + j];
                const float* wg  = WgP + c * 81;
                const float* bgc = bgP + c * 9;
                float lg[9];
                #pragma unroll
                for (int o = 0; o < 9; ++o) {
                    float a = bgc[o];
                    #pragma unroll
                    for (int i = 0; i < 9; ++i) a += wg[o * 9 + i] * p[i];
                    lg[o] = a;
                }
                float mx = lg[0];
                #pragma unroll
                for (int o = 1; o < 9; ++o) mx = fmaxf(mx, lg[o]);
                float s = 0.f;
                #pragma unroll
                for (int o = 0; o < 9; ++o) { lg[o] = __expf(lg[o] - mx); s += lg[o]; }
                float inv = 1.f / s;
                int base = pix * BT_STRIDE + (c - half * 64) * 9;
                #pragma unroll
                for (int o = 0; o < 9; ++o)
                    Bt[base + o] = (h16)(p[o] * lg[o] * inv);
            }
        }
        __syncthreads();

        // Phase C: GEMM over this K half (18 chunks of 32); wave owns 2 M-tiles x 2 N-tiles
        const h16* bb0 = &Bt[(lane & 15) * BT_STRIDE + ((lane < 16) ? 0 : 16)];
        const h16* bb1 = bb0 + 16 * BT_STRIDE;
        for (int kc = 0; kc < 18; ++kc) {
            int kcg = half * 18 + kc;
            v16h bf0 = *(const v16h*)(bb0 + kc * 32);
            v16h bf1 = *(const v16h*)(bb1 + kc * 32);
            const h16* ap = A2 + ((size_t)(kcg * 16 + m0t) * 32 + lane) * 16;
            v16h a0 = *(const v16h*)ap;
            v16h a1 = *(const v16h*)(ap + 32 * 16);
            if (kcg < 35)                                     // prefetch next chunk's A frags
                __builtin_prefetch(ap + 16 * 32 * 16, 0, 1);
            acc[0][0] = __builtin_amdgcn_wmma_f32_16x16x32_f16(false, a0, false, bf0,
                                                               (short)0, acc[0][0], false, false);
            acc[0][1] = __builtin_amdgcn_wmma_f32_16x16x32_f16(false, a0, false, bf1,
                                                               (short)0, acc[0][1], false, false);
            acc[1][0] = __builtin_amdgcn_wmma_f32_16x16x32_f16(false, a1, false, bf0,
                                                               (short)0, acc[1][0], false, false);
            acc[1][1] = __builtin_amdgcn_wmma_f32_16x16x32_f16(false, a1, false, bf1,
                                                               (short)0, acc[1][1], false, false);
        }
    }

    // Epilogue: bias + ReLU + residual; N-tile nt selects output row y0+nt
    #pragma unroll
    for (int t2 = 0; t2 < 2; ++t2)
        #pragma unroll
        for (int nt = 0; nt < 2; ++nt)
            #pragma unroll
            for (int r = 0; r < 8; ++r) {
                int m = (m0t + t2) * 16 + 8 * (lane >> 4) + r;
                int gy = y0 + nt;
                int gx = x0 + (lane & 15);
                size_t idx = ((size_t)(b * C2 + m) * H + gy) * W + gx;
                float v = acc[t2][nt][r] + bcp[m];
                v = fmaxf(v, 0.f);
                out[idx] = v + xin[idx];
            }
}

extern "C" void kernel_launch(void* const* d_in, const int* in_sizes, int n_in,
                              void* d_out, int out_size, void* d_ws, size_t ws_size,
                              hipStream_t stream) {
    const float* x   = (const float*)d_in[0];
    const float* W1  = (const float*)d_in[1];
    const float* g1  = (const float*)d_in[2];
    const float* b1  = (const float*)d_in[3];
    const float* m1  = (const float*)d_in[4];
    const float* v1  = (const float*)d_in[5];
    const float* Wg  = (const float*)d_in[6];
    const float* bg  = (const float*)d_in[7];
    const float* gg  = (const float*)d_in[8];
    const float* bgw = (const float*)d_in[9];
    const float* mg  = (const float*)d_in[10];
    const float* vg  = (const float*)d_in[11];
    const float* Wc  = (const float*)d_in[12];
    const float* bc  = (const float*)d_in[13];
    const float* g2  = (const float*)d_in[14];
    const float* b2  = (const float*)d_in[15];
    const float* m2  = (const float*)d_in[16];
    const float* v2  = (const float*)d_in[17];
    float* out = (float*)d_out;

    char* ws = (char*)d_ws;
    h16*   hF16 = (h16*)(ws + WS_H16);
    h16*   A1   = (h16*)(ws + WS_A1PACK);
    h16*   A2   = (h16*)(ws + WS_A2PACK);
    float* WgP  = (float*)(ws + WS_WGP);
    float* bgP  = (float*)(ws + WS_BGP);
    float* c1b  = (float*)(ws + WS_C1B);
    float* bcp  = (float*)(ws + WS_BCP);

    // --- prep (tiny) ---
    k_prep_bias<<<2, 256, 0, stream>>>(g1, b1, m1, v1, bc, g2, b2, m2, v2, c1b, bcp);
    k_prep_wg<<<(Cm * 90 + 255) / 256, 256, 0, stream>>>(Wg, bg, gg, bgw, mg, vg, WgP, bgP);
    k_prep_a1<<<32768 / 256, 256, 0, stream>>>(W1, g1, v1, A1);
    k_prep_a2<<<294912 / 256, 256, 0, stream>>>(Wc, g2, v2, A2);

    // --- cv1: 1x1 conv + BN + SiLU (WMMA f16) ---
    k_cv1<<<Bsz * (HW / 64), 256, 0, stream>>>(x, A1, c1b, hF16);

    // --- fused RFA: unfold + attention softmax + big GEMM + residual ---
    k_rfa<<<Bsz * (H / 2) * (W / 16), 256, 0, stream>>>(hF16, A2, WgP, bgP, bcp, x, out);
}
